// GapLoss_41601053229217
// MI455X (gfx1250) — matrix operand order repaired and checked by reading, not compiled, not run on hardware
//
#include <hip/hip_runtime.h>
#include <hip/hip_bf16.h>
#include <stdint.h>

typedef __attribute__((ext_vector_type(16))) _Float16 v16h;
typedef __attribute__((ext_vector_type(8)))  float    v8f;

// Problem dims (fixed by the reference)
#define BB   16
#define NP   4
#define HH   512
#define WW   512
#define NSL  (BB*NP)          // 64 slices
#define WPR  (WW/32)          // 16 packed words per row
#define WORDS_PER_SLICE (HH*WPR)   // 8192
#define NPHW (NP*HH*WW)       // 1,048,576
#define NTOT ((size_t)BB*NP*HH*WW) // 16,777,216

// Workspace layout (bytes)
#define AB_OFF  ((size_t)0)                          // skel bits  : 2 MB
#define CB_OFF  ((size_t)(2u<<20))                   // endpoint C : 2 MB
#define HS_OFF  ((size_t)(4u<<20))                   // f16 hsum   : 32 MB
#define WM_OFF  (HS_OFF + (size_t)NSL*HH*WW*2)       // u8 wmap    : 16 MB

// ---------------------------------------------------------------------------
// K0: logsumexp over axis-0, threshold, ballot-pack into bitboards
// ---------------------------------------------------------------------------
__global__ void k_binarize(const float* __restrict__ x, uint32_t* __restrict__ Abits) {
    int idx = blockIdx.x * blockDim.x + threadIdx.x;     // over NP*H*W, full waves
    float xv[BB];
    float m = -3.4e38f;
#pragma unroll
    for (int b = 0; b < BB; ++b) {
        xv[b] = x[(size_t)b * NPHW + idx];
        m = fmaxf(m, xv[b]);
    }
    float s = 0.f;
#pragma unroll
    for (int b = 0; b < BB; ++b) s += __expf(xv[b] - m);
    float thr = m + __logf(s) - 0.69314718056f;          // lse - ln2

    int p = idx / (HH * WW);
    int rem = idx - p * (HH * WW);
    int h = rem >> 9;
    int w = rem & 511;
    int lane = threadIdx.x & 31;
#pragma unroll
    for (int b = 0; b < BB; ++b) {
        unsigned long long bal = __ballot(xv[b] >= thr); // wave32: low 32 bits
        if (lane == 0)
            Abits[(size_t)(b * NP + p) * WORDS_PER_SLICE + h * WPR + (w >> 5)] =
                (uint32_t)bal;
    }
}

// ---------------------------------------------------------------------------
// Bitboard helpers: bit i of a word = pixel at column base+i.
// East neighbor board (pixel col+1): shift toward LSB, pull from next word.
// ---------------------------------------------------------------------------
__device__ __forceinline__ uint32_t shE(uint32_t l, uint32_t c, uint32_t r) {
    return (c >> 1) | (r << 31);
}
__device__ __forceinline__ uint32_t shW(uint32_t l, uint32_t c, uint32_t r) {
    return (c << 1) | (l >> 31);
}
// carry-save add of a 1-bit board into a 4-bit bit-sliced accumulator
__device__ __forceinline__ void csadd(uint32_t b, uint32_t& s0, uint32_t& s1,
                                      uint32_t& s2, uint32_t& s3) {
    uint32_t c0 = s0 & b;  s0 ^= b;
    uint32_t c1 = s1 & c0; s1 ^= c0;
    uint32_t c2 = s2 & c1; s2 ^= c1;
    s3 ^= c2;
}

// ---------------------------------------------------------------------------
// K1: 16 Zhang–Suen substeps, one workgroup per slice, slice lives in LDS.
// 512 threads, one row per thread, bit-sliced neighbor/transition counting.
// Early-out on all-zero center words (data is very sparse:
// softmax over 16 N(0,1) logits rarely reaches 0.5).
// ---------------------------------------------------------------------------
__global__ __launch_bounds__(512) void k_thin(uint32_t* __restrict__ Abits) {
    __shared__ uint32_t img[HH][WPR + 1];                // +1 pad: 64-bank safe
    int s = blockIdx.x;
    int r = threadIdx.x;
    uint32_t* g = Abits + (size_t)s * WORDS_PER_SLICE;
#pragma unroll
    for (int w = 0; w < WPR; ++w) img[r][w] = g[r * WPR + w];
    __syncthreads();

    for (int step = 0; step < 16; ++step) {
        int sub = step & 1;
        uint32_t nw[WPR];
#pragma unroll
        for (int w = 0; w < WPR; ++w) {
            uint32_t cc = img[r][w];
            uint32_t cl = (w > 0)       ? img[r][w - 1] : 0u;
            uint32_t cr = (w < WPR - 1) ? img[r][w + 1] : 0u;
            uint32_t uc = 0, ul = 0, ur = 0, dc = 0, dl = 0, dr = 0;
            if (r > 0) {
                uc = img[r - 1][w];
                ul = (w > 0)       ? img[r - 1][w - 1] : 0u;
                ur = (w < WPR - 1) ? img[r - 1][w + 1] : 0u;
            }
            if (r < HH - 1) {
                dc = img[r + 1][w];
                dl = (w > 0)       ? img[r + 1][w - 1] : 0u;
                dr = (w < WPR - 1) ? img[r + 1][w + 1] : 0u;
            }
            if (cc == 0u) {                  // no set pixels -> nothing to delete
                nw[w] = 0u;
                continue;
            }
            // p2..p9 clockwise: N, NE, E, SE, S, SW, W, NW
            uint32_t n0 = uc, n1 = shE(ul, uc, ur), n2 = shE(cl, cc, cr),
                     n3 = shE(dl, dc, dr), n4 = dc, n5 = shW(dl, dc, dr),
                     n6 = shW(cl, cc, cr), n7 = shW(ul, uc, ur);
            // Bn = neighbor count (bit-sliced 4-bit)
            uint32_t s0 = n0, s1 = 0, s2 = 0, s3 = 0;
            csadd(n1, s0, s1, s2, s3); csadd(n2, s0, s1, s2, s3);
            csadd(n3, s0, s1, s2, s3); csadd(n4, s0, s1, s2, s3);
            csadd(n5, s0, s1, s2, s3); csadd(n6, s0, s1, s2, s3);
            csadd(n7, s0, s1, s2, s3);
            uint32_t ge2 = s1 | s2 | s3;
            uint32_t le6 = ~(s3 | (s2 & s1 & s0));
            // transitions: count of (0->1) around ring, need == 1
            uint32_t t0 = (~n0) & n1, t1 = (~n1) & n2, t2 = (~n2) & n3,
                     t3 = (~n3) & n4, t4 = (~n4) & n5, t5 = (~n5) & n6,
                     t6 = (~n6) & n7, t7 = (~n7) & n0;
            uint32_t u0 = t0, u1 = 0, u2 = 0, u3 = 0;
            csadd(t1, u0, u1, u2, u3); csadd(t2, u0, u1, u2, u3);
            csadd(t3, u0, u1, u2, u3); csadd(t4, u0, u1, u2, u3);
            csadd(t5, u0, u1, u2, u3); csadd(t6, u0, u1, u2, u3);
            csadd(t7, u0, u1, u2, u3);
            uint32_t tr1 = u0 & ~u1 & ~u2;
            uint32_t c1 = ge2 & le6 & tr1;
            uint32_t c2 = sub == 0
                              ? (~(n0 & n2 & n4) & ~(n2 & n4 & n6))
                              : (~(n0 & n2 & n6) & ~(n0 & n4 & n6));
            nw[w] = cc & ~(c1 & c2);
        }
        __syncthreads();
#pragma unroll
        for (int w = 0; w < WPR; ++w) img[r][w] = nw[w];
        __syncthreads();
    }
#pragma unroll
    for (int w = 0; w < WPR; ++w) g[r * WPR + w] = img[r][w];
}

// ---------------------------------------------------------------------------
// K2: endpoint map. conv3x3(zero-center) >= 8  <=>  AND of all 8 neighbors.
// ---------------------------------------------------------------------------
__global__ void k_endpoint(const uint32_t* __restrict__ Sk,
                           uint32_t* __restrict__ Cb) {
    int gid = blockIdx.x * blockDim.x + threadIdx.x;     // over NSL*H*WPR
    int s = gid / WORDS_PER_SLICE;
    int rem = gid - s * WORDS_PER_SLICE;
    int r = rem / WPR;
    int w = rem - r * WPR;
    const uint32_t* g = Sk + (size_t)s * WORDS_PER_SLICE;
    uint32_t out = 0;
    if (r > 0 && r < HH - 1) {
        uint32_t uc = g[(r - 1) * WPR + w];
        uint32_t dc = g[(r + 1) * WPR + w];
        uint32_t both = uc & dc;
        if (both) {                                      // sparse early-out
            uint32_t cc = g[r * WPR + w];
            uint32_t cl = (w > 0)       ? g[r * WPR + w - 1] : 0u;
            uint32_t cr = (w < WPR - 1) ? g[r * WPR + w + 1] : 0u;
            uint32_t ul = (w > 0)       ? g[(r - 1) * WPR + w - 1] : 0u;
            uint32_t ur = (w < WPR - 1) ? g[(r - 1) * WPR + w + 1] : 0u;
            uint32_t dl = (w > 0)       ? g[(r + 1) * WPR + w - 1] : 0u;
            uint32_t dr = (w < WPR - 1) ? g[(r + 1) * WPR + w + 1] : 0u;
            out = both & shE(ul, uc, ur) & shE(cl, cc, cr) & shE(dl, dc, dr) &
                  shW(ul, uc, ur) & shW(cl, cc, cr) & shW(dl, dc, dr);
        }
    }
    Cb[gid] = out;
}

// band(k,n): ones matrix for width-9 sliding window centered 8 deep in K
__device__ __forceinline__ _Float16 bandval(int k, int n) {
    return (k >= n + 4 && k <= n + 12) ? (_Float16)1.0f : (_Float16)0.0f;
}

// ---------------------------------------------------------------------------
// K3a: horizontal width-9 window sum via V_WMMA_F32_16X16X32_F16.
// One wave per 16x16 output tile. A = C bits (16x32 f16), B = band ones.
// ---------------------------------------------------------------------------
__global__ void k_hconv(const uint32_t* __restrict__ Cb,
                        _Float16* __restrict__ Hs) {
    int wave = (blockIdx.x * blockDim.x + threadIdx.x) >> 5;
    int lane = threadIdx.x & 31;
    int slice = wave >> 10;
    int t = wave & 1023;
    int r0 = (t >> 5) * 16, c0 = (t & 31) * 16;
    int half = lane >> 4, m = lane & 15;

    const uint32_t* crow = Cb + (size_t)slice * WORDS_PER_SLICE + (r0 + m) * WPR;
    v16h a;
#pragma unroll
    for (int v = 0; v < 8; ++v) {
        int kb = 2 * v + (v >= 4 ? 8 : 0) + half * 8;    // A 16x32 f16 layout
#pragma unroll
        for (int e = 0; e < 2; ++e) {
            int col = c0 - 8 + kb + e;
            float val = 0.f;
            if (col >= 0 && col < WW)
                val = (float)((crow[col >> 5] >> (col & 31)) & 1u);
            a[2 * v + e] = (_Float16)val;
        }
    }
    v16h bmat;
#pragma unroll
    for (int v = 0; v < 8; ++v) {                        // B 32x16 f16 layout
        int k0 = 2 * v + (v >= 4 ? 8 : 0) + half * 8;
        bmat[2 * v]     = bandval(k0, m);
        bmat[2 * v + 1] = bandval(k0 + 1, m);
    }
    v8f acc = {};
    acc = __builtin_amdgcn_wmma_f32_16x16x32_f16(false, a, false, bmat,
                                                 (short)0, acc, false, false);
    _Float16* hsl = Hs + (size_t)slice * HH * WW;
#pragma unroll
    for (int i = 0; i < 8; ++i) {                        // D: M=i(+8), N=lane%16
        int rr = r0 + i + half * 8;
        hsl[rr * WW + c0 + m] = (_Float16)acc[i];
    }
}

// ---------------------------------------------------------------------------
// K3b: vertical width-9 window sum (transposed fragment), subtract center C,
// emit u8 weight counts (<= 80, exact).
// ---------------------------------------------------------------------------
__global__ void k_vconv(const _Float16* __restrict__ Hs,
                        const uint32_t* __restrict__ Cb,
                        uint8_t* __restrict__ Wm) {
    int wave = (blockIdx.x * blockDim.x + threadIdx.x) >> 5;
    int lane = threadIdx.x & 31;
    int slice = wave >> 10;
    int t = wave & 1023;
    int r0 = (t >> 5) * 16, c0 = (t & 31) * 16;
    int half = lane >> 4, m = lane & 15;

    const _Float16* hsl = Hs + (size_t)slice * HH * WW;
    v16h a;                                              // A[M=c, K=row offset]
#pragma unroll
    for (int v = 0; v < 8; ++v) {
        int kb = 2 * v + (v >= 4 ? 8 : 0) + half * 8;
#pragma unroll
        for (int e = 0; e < 2; ++e) {
            int row = r0 - 8 + kb + e;
            _Float16 val = (_Float16)0.0f;
            if (row >= 0 && row < HH) val = hsl[row * WW + c0 + m];
            a[2 * v + e] = val;
        }
    }
    v16h bmat;
#pragma unroll
    for (int v = 0; v < 8; ++v) {
        int k0 = 2 * v + (v >= 4 ? 8 : 0) + half * 8;
        bmat[2 * v]     = bandval(k0, m);
        bmat[2 * v + 1] = bandval(k0 + 1, m);
    }
    v8f acc = {};
    acc = __builtin_amdgcn_wmma_f32_16x16x32_f16(false, a, false, bmat,
                                                 (short)0, acc, false, false);
    const uint32_t* csl = Cb + (size_t)slice * WORDS_PER_SLICE;
    uint8_t* wsl = Wm + (size_t)slice * HH * WW;
#pragma unroll
    for (int i = 0; i < 8; ++i) {                        // D: M=c-idx, N=row-idx
        int col = c0 + i + half * 8;
        int row = r0 + m;
        uint32_t cbit = (csl[row * WPR + (col >> 5)] >> (col & 31)) & 1u;
        int v = (int)(acc[i] + 0.5f) - (int)cbit;        // full 9x9 sum - center
        wsl[row * WW + col] = (uint8_t)v;
    }
}

// ---------------------------------------------------------------------------
// K4: recompute BCE (coalesced x,y re-read), multiply by weight, reduce.
// Loads of x,y are predicated on nonzero weight: whole cachelines of the
// 134 MB re-read are skipped wherever the (sparse) weight map is zero.
// ---------------------------------------------------------------------------
__global__ void k_reduce(const float* __restrict__ x, const float* __restrict__ y,
                         const uint8_t* __restrict__ Wm, float* __restrict__ out) {
    __shared__ float buf[256];
    size_t stride = (size_t)gridDim.x * blockDim.x;
    float acc = 0.f;
    for (size_t i = (size_t)blockIdx.x * blockDim.x + threadIdx.x; i < NTOT;
         i += stride) {
        uint8_t w = Wm[i];
        if (w) {
            float xv = x[i], yv = y[i];
            float l = fmaxf(xv, 0.f) - xv * yv + log1pf(__expf(-fabsf(xv)));
            acc += l * (float)w;
        }
    }
    buf[threadIdx.x] = acc;
    __syncthreads();
    for (int off = 128; off > 0; off >>= 1) {
        if (threadIdx.x < off) buf[threadIdx.x] += buf[threadIdx.x + off];
        __syncthreads();
    }
    if (threadIdx.x == 0)
        atomicAdd(out, buf[0] * (60.0f / (float)NTOT));  // fold K_WEIGHT & mean
}

// ---------------------------------------------------------------------------
extern "C" void kernel_launch(void* const* d_in, const int* in_sizes, int n_in,
                              void* d_out, int out_size, void* d_ws, size_t ws_size,
                              hipStream_t stream) {
    const float* x = (const float*)d_in[0];
    const float* y = (const float*)d_in[1];
    float* out = (float*)d_out;
    char* ws = (char*)d_ws;

    uint32_t* Abits = (uint32_t*)(ws + AB_OFF);
    uint32_t* Cbits = (uint32_t*)(ws + CB_OFF);
    _Float16* Hsum  = (_Float16*)(ws + HS_OFF);
    uint8_t*  Wmap  = (uint8_t*)(ws + WM_OFF);

    hipMemsetAsync(out, 0, sizeof(float), stream);

    k_binarize<<<NPHW / 256, 256, 0, stream>>>(x, Abits);
    k_thin<<<NSL, 512, 0, stream>>>(Abits);
    k_endpoint<<<(NSL * WORDS_PER_SLICE) / 256, 256, 0, stream>>>(Abits, Cbits);
    // 64 slices * 32*32 tiles = 65536 waves; 8 waves per 256-thread block
    k_hconv<<<65536 / 8, 256, 0, stream>>>(Cbits, Hsum);
    k_vconv<<<65536 / 8, 256, 0, stream>>>(Hsum, Cbits, Wmap);
    k_reduce<<<4096, 256, 0, stream>>>(x, y, Wmap, out);
}
